// model_77824807404112
// MI455X (gfx1250) — 8-bit matrix operand order repaired and checked by reading, compiled, not run on hardware
//
#include <hip/hip_runtime.h>
#include <cstdint>

typedef int v8i __attribute__((ext_vector_type(8)));

static const size_t OFF_ACC  = 0;
static const size_t OFF_S    = 64;
static const size_t OFF_QX   = 256;
static const size_t OFF_ACT1 = OFF_QX   + 802816;
static const size_t OFF_QW1  = OFF_ACT1 + 5537792;
static const size_t OFF_QW2P = OFF_QW1  + 512;
static const size_t OFF_QWL1 = OFF_QW2P + 20480;
static const size_t OFF_QWL2 = OFF_QWL1 + 6553600;
static const size_t OFF_ACT2 = OFF_QWL2 + 40960;
static const size_t OFF_ACT3 = OFF_ACT2 + 1638400;
static const size_t OFF_RAW3 = OFF_ACT3 + 4194304;

__device__ __forceinline__ int dot4_i8(int a, int b) {
  return (int)(signed char)(a)        * (int)(signed char)(b)
       + (int)(signed char)(a >> 8)   * (int)(signed char)(b >> 8)
       + (int)(signed char)(a >> 16)  * (int)(signed char)(b >> 16)
       + (a >> 24)                    * (b >> 24);
}

__device__ __forceinline__ float clampf(float v, float lo, float hi) {
  return fminf(fmaxf(v, lo), hi);
}

__global__ void k_init(unsigned* acc) {
  if (threadIdx.x < 4) acc[threadIdx.x] = 0u;
}

__global__ void k_absmax(const float* __restrict__ w, int n, unsigned* __restrict__ acc) {
  __shared__ unsigned red[256];
  unsigned m = 0u;
  for (int i = blockIdx.x * blockDim.x + threadIdx.x; i < n; i += gridDim.x * blockDim.x) {
    unsigned v = __float_as_uint(fabsf(w[i]));
    m = (v > m) ? v : m;
  }
  red[threadIdx.x] = m;
  __syncthreads();
  for (int s = 128; s > 0; s >>= 1) {
    if ((int)threadIdx.x < s) {
      unsigned o = red[threadIdx.x + s];
      if (o > red[threadIdx.x]) red[threadIdx.x] = o;
    }
    __syncthreads();
  }
  if (threadIdx.x == 0) atomicMax(acc, red[0]);
}

__global__ void k_quant_x(const float* __restrict__ x, const float* __restrict__ s_in_p,
                          int8_t* __restrict__ q, int n) {
  int i = blockIdx.x * blockDim.x + threadIdx.x;
  if (i >= n) return;
  float s = s_in_p[0];
  float r = clampf(rintf(x[i] / s), -8.0f, 7.0f);
  q[i] = (int8_t)(int)r;
}

__global__ void k_quant_w(const float* __restrict__ w, int n,
                          const unsigned* __restrict__ amax,
                          float* __restrict__ s_out, int8_t* __restrict__ q) {
  float s = __uint_as_float(*amax) * (1.0f / 7.0f);
  int i = blockIdx.x * blockDim.x + threadIdx.x;
  if (i == 0) *s_out = s;
  if (i >= n) return;
  float r = clampf(rintf(w[i] / s), -7.0f, 7.0f);
  q[i] = (int8_t)(int)r;
}

__global__ void k_quant_w2p(const float* __restrict__ w, const unsigned* __restrict__ amax,
                            float* __restrict__ s_out, int8_t* __restrict__ q) {
  float s = __uint_as_float(*amax) * (1.0f / 7.0f);
  int i = blockIdx.x * blockDim.x + threadIdx.x;
  if (i == 0) *s_out = s;
  if (i >= 64 * 320) return;
  int kk = i % 320, o = i / 320;
  int8_t qv = 0;
  if (kk < 288) {
    int ci = kk & 31, t = kk >> 5;
    int ky = t / 3, kx = t - ky * 3;
    float wv = w[((o * 32 + ci) * 3 + ky) * 3 + kx];
    float r = clampf(rintf(wv / s), -7.0f, 7.0f);
    qv = (int8_t)(int)r;
  }
  q[i] = qv;
}

__global__ void k_conv1(const int8_t* __restrict__ qx, const int8_t* __restrict__ qw1,
                        const float* __restrict__ sbuf, const float* __restrict__ s_in_p,
                        const float* __restrict__ s_a1_p, int8_t* __restrict__ act1) {
  int tid = blockIdx.x * blockDim.x + threadIdx.x;
  int c = tid & 31; int t = tid >> 5;
  int px = t % 13; t /= 13;
  int py = t % 13; int b = t / 13;

  int w[9];
#pragma unroll
  for (int i = 0; i < 9; i++) w[i] = qw1[c * 9 + i];

  float s_in = s_in_p[0], s_a1 = s_a1_p[0];
  float cscale = s_in * sbuf[0];
  const int8_t* xb = qx + b * 784;

  float rmax = 0.0f;
#pragma unroll
  for (int dy = 0; dy < 2; dy++) {
#pragma unroll
    for (int dx = 0; dx < 2; dx++) {
      int oy = 2 * py + dy, ox = 2 * px + dx;
      int acc = 0;
#pragma unroll
      for (int ky = 0; ky < 3; ky++)
#pragma unroll
        for (int kx = 0; kx < 3; kx++)
          acc += (int)xb[(oy + ky) * 28 + (ox + kx)] * w[ky * 3 + kx];
      float r = clampf(rintf(((float)acc * cscale) / s_a1), 0.0f, 15.0f);
      rmax = fmaxf(rmax, r);
    }
  }
  float qv = clampf(rintf(rmax * s_a1 / s_in), -8.0f, 7.0f);
  act1[tid] = (int8_t)(int)qv;
}

__global__ void __launch_bounds__(256) k_conv2_wmma(
    const int8_t* __restrict__ act1, const int8_t* __restrict__ Wp,
    const float* __restrict__ sbuf, const float* __restrict__ s_in_p,
    const float* __restrict__ s_a2_p, int8_t* __restrict__ raw) {
  int wave = (blockIdx.x * blockDim.x + threadIdx.x) >> 5;
  int lane = threadIdx.x & 31;
  int r = lane & 15;
  int h = lane >> 4;

  int m0 = wave << 4;
  int m = m0 + r;
  int ox = m % 11; int t = m / 11;
  int oy = t % 11; int b = t / 11;
  const int8_t* abase = act1 + ((size_t)(b * 13 + oy) * 13 + ox) * 32;

  const int8_t* w0 = Wp + (size_t)(0 * 16 + r) * 320;
  const int8_t* w1 = Wp + (size_t)(1 * 16 + r) * 320;
  const int8_t* w2 = Wp + (size_t)(2 * 16 + r) * 320;
  const int8_t* w3 = Wp + (size_t)(3 * 16 + r) * 320;

  v8i c0 = {0,0,0,0,0,0,0,0}, c1 = {0,0,0,0,0,0,0,0};
  v8i c2 = {0,0,0,0,0,0,0,0}, c3 = {0,0,0,0,0,0,0,0};

#pragma unroll
  for (int s = 0; s < 5; s++) {
    v8i a;
#pragma unroll
    for (int v = 0; v < 8; v++) {
      int kk = s * 64 + ((v >> 1) << 4) + (h << 3) + ((v & 1) << 2);
      int val = 0;
      if (kk < 288) {
        int tap = kk >> 5, ci = kk & 31;
        int ky = tap / 3, kx = tap - ky * 3;
        val = *(const int*)(abase + (ky * 13 + kx) * 32 + ci);
      }
      a[v] = val;
    }
    v8i b0, b1, b2, b3;
#pragma unroll
    for (int v = 0; v < 8; v++) {
      int kb = s * 64 + ((v >> 1) << 4) + (h << 3) + ((v & 1) << 2);
      b0[v] = *(const int*)(w0 + kb);
      b1[v] = *(const int*)(w1 + kb);
      b2[v] = *(const int*)(w2 + kb);
      b3[v] = *(const int*)(w3 + kb);
    }
    c0 = __builtin_amdgcn_wmma_i32_16x16x64_iu8(true, a, true, b0, c0, false, false);
    c1 = __builtin_amdgcn_wmma_i32_16x16x64_iu8(true, a, true, b1, c1, false, false);
    c2 = __builtin_amdgcn_wmma_i32_16x16x64_iu8(true, a, true, b2, c2, false, false);
    c3 = __builtin_amdgcn_wmma_i32_16x16x64_iu8(true, a, true, b3, c3, false, false);
  }

  float cscale = s_in_p[0] * sbuf[1];
  float s_a2 = s_a2_p[0];
  int N = lane & 15;
#pragma unroll
  for (int rr = 0; rr < 8; rr++) {
    int M = rr + (h << 3);
    size_t row = (size_t)(m0 + M) * 64;
    float q0 = clampf(rintf(((float)c0[rr] * cscale) / s_a2), 0.0f, 15.0f);
    float q1 = clampf(rintf(((float)c1[rr] * cscale) / s_a2), 0.0f, 15.0f);
    float q2 = clampf(rintf(((float)c2[rr] * cscale) / s_a2), 0.0f, 15.0f);
    float q3 = clampf(rintf(((float)c3[rr] * cscale) / s_a2), 0.0f, 15.0f);
    raw[row +  0 + N] = (int8_t)(int)q0;
    raw[row + 16 + N] = (int8_t)(int)q1;
    raw[row + 32 + N] = (int8_t)(int)q2;
    raw[row + 48 + N] = (int8_t)(int)q3;
  }
}

__global__ void k_pool2(const int8_t* __restrict__ raw,
                        const float* __restrict__ s_in_p, const float* __restrict__ s_a2_p,
                        int8_t* __restrict__ act2) {
  int tid = blockIdx.x * blockDim.x + threadIdx.x;
  int px = tid % 5; int t = tid / 5;
  int py = t % 5; t /= 5;
  int c = t % 64; int b = t / 64;
  int rmax = 0;
#pragma unroll
  for (int dy = 0; dy < 2; dy++)
#pragma unroll
    for (int dx = 0; dx < 2; dx++) {
      int oy = 2 * py + dy, ox = 2 * px + dx;
      int v = raw[((size_t)(b * 11 + oy) * 11 + ox) * 64 + c];
      rmax = v > rmax ? v : rmax;
    }
  float s_in = s_in_p[0], s_a2 = s_a2_p[0];
  float qv = clampf(rintf((float)rmax * s_a2 / s_in), -8.0f, 7.0f);
  act2[(size_t)b * 1600 + c * 25 + py * 5 + px] = (int8_t)(int)qv;
}

__global__ void __launch_bounds__(256) k_fc1_wmma(
    const int8_t* __restrict__ A, const int8_t* __restrict__ W,
    const float* __restrict__ sbuf, const float* __restrict__ s_in_p,
    const float* __restrict__ s_a3_p, int8_t* __restrict__ R3) {
  int wave = (blockIdx.x * blockDim.x + threadIdx.x) >> 5;
  int lane = threadIdx.x & 31;
  int tnq = wave & 63;
  int tm = wave >> 6;
  int m0 = tm << 4, n0 = tnq << 6;
  int r = lane & 15;
  int h = lane >> 4;

  const int8_t* arow = A + (size_t)(m0 + r) * 1600;
  const int8_t* w0 = W + (size_t)(n0 +  0 + r) * 1600;
  const int8_t* w1 = W + (size_t)(n0 + 16 + r) * 1600;
  const int8_t* w2 = W + (size_t)(n0 + 32 + r) * 1600;
  const int8_t* w3 = W + (size_t)(n0 + 48 + r) * 1600;

  v8i c0 = {0,0,0,0,0,0,0,0}, c1 = {0,0,0,0,0,0,0,0};
  v8i c2 = {0,0,0,0,0,0,0,0}, c3 = {0,0,0,0,0,0,0,0};

  for (int k0 = 0; k0 < 1600; k0 += 64) {
    __builtin_prefetch(arow + k0 + 128, 0, 3);
    __builtin_prefetch(w0 + k0 + 128, 0, 3);
    __builtin_prefetch(w1 + k0 + 128, 0, 3);
    __builtin_prefetch(w2 + k0 + 128, 0, 3);
    __builtin_prefetch(w3 + k0 + 128, 0, 3);
    v8i a, b0, b1, b2, b3;
#pragma unroll
    for (int v = 0; v < 8; v++) {
      int ka = ((v >> 1) << 4) + (h << 3) + ((v & 1) << 2);
      a[v] = *(const int*)(arow + k0 + ka);
    }
#pragma unroll
    for (int v = 0; v < 8; v++) {
      int kb = ((v >> 1) << 4) + (h << 3) + ((v & 1) << 2);
      b0[v] = *(const int*)(w0 + k0 + kb);
      b1[v] = *(const int*)(w1 + k0 + kb);
      b2[v] = *(const int*)(w2 + k0 + kb);
      b3[v] = *(const int*)(w3 + k0 + kb);
    }
    c0 = __builtin_amdgcn_wmma_i32_16x16x64_iu8(true, a, true, b0, c0, false, false);
    c1 = __builtin_amdgcn_wmma_i32_16x16x64_iu8(true, a, true, b1, c1, false, false);
    c2 = __builtin_amdgcn_wmma_i32_16x16x64_iu8(true, a, true, b2, c2, false, false);
    c3 = __builtin_amdgcn_wmma_i32_16x16x64_iu8(true, a, true, b3, c3, false, false);
  }

  float scale = s_in_p[0] * sbuf[2];
  float s_a3 = s_a3_p[0];
  int N = lane & 15;
#pragma unroll
  for (int rr = 0; rr < 8; rr++) {
    int M = rr + (h << 3);
    size_t row = (size_t)(m0 + M) * 4096 + n0;
    float q0 = clampf(rintf(((float)c0[rr] * scale) / s_a3), 0.0f, 15.0f);
    float q1 = clampf(rintf(((float)c1[rr] * scale) / s_a3), 0.0f, 15.0f);
    float q2 = clampf(rintf(((float)c2[rr] * scale) / s_a3), 0.0f, 15.0f);
    float q3 = clampf(rintf(((float)c3[rr] * scale) / s_a3), 0.0f, 15.0f);
    R3[row +  0 + N] = (int8_t)(int)q0;
    R3[row + 16 + N] = (int8_t)(int)q1;
    R3[row + 32 + N] = (int8_t)(int)q2;
    R3[row + 48 + N] = (int8_t)(int)q3;
  }
}

__global__ void k_fc2(const int8_t* __restrict__ R3, const int8_t* __restrict__ qwl2,
                      const float* __restrict__ sbuf, const float* __restrict__ s_a3_p,
                      float* __restrict__ out) {
  int tid = blockIdx.x * blockDim.x + threadIdx.x;
  if (tid >= 1024 * 10) return;
  int j = tid % 10, b = tid / 10;
  const int* xp = (const int*)(R3 + (size_t)b * 4096);
  const int* wp = (const int*)(qwl2 + (size_t)j * 4096);
  int acc = 0;
  for (int k = 0; k < 1024; k++) acc += dot4_i8(xp[k], wp[k]);
  out[tid] = (float)acc * (s_a3_p[0] * sbuf[3]);
}

extern "C" void kernel_launch(void* const* d_in, const int* in_sizes, int n_in,
                              void* d_out, int out_size, void* d_ws, size_t ws_size,
                              hipStream_t stream) {
  const float* x    = (const float*)d_in[0];
  const float* w1   = (const float*)d_in[1];
  const float* w2   = (const float*)d_in[2];
  const float* wl1  = (const float*)d_in[3];
  const float* wl2  = (const float*)d_in[4];
  const float* s_in = (const float*)d_in[5];
  const float* s_a1 = (const float*)d_in[6];
  const float* s_a2 = (const float*)d_in[7];
  const float* s_a3 = (const float*)d_in[8];

  char* ws = (char*)d_ws;
  unsigned* acc  = (unsigned*)(ws + OFF_ACC);
  float*    sbuf = (float*)(ws + OFF_S);
  int8_t* qx   = (int8_t*)(ws + OFF_QX);
  int8_t* act1 = (int8_t*)(ws + OFF_ACT1);
  int8_t* qw1  = (int8_t*)(ws + OFF_QW1);
  int8_t* qw2p = (int8_t*)(ws + OFF_QW2P);
  int8_t* qwl1 = (int8_t*)(ws + OFF_QWL1);
  int8_t* qwl2 = (int8_t*)(ws + OFF_QWL2);
  int8_t* act2 = (int8_t*)(ws + OFF_ACT2);
  int8_t* act3 = (int8_t*)(ws + OFF_ACT3);
  int8_t* raw3 = (int8_t*)(ws + OFF_RAW3);

  k_init<<<1, 32, 0, stream>>>(acc);
  k_absmax<<<8,   256, 0, stream>>>(w1,  288,         acc + 0);
  k_absmax<<<32,  256, 0, stream>>>(w2,  18432,       acc + 1);
  k_absmax<<<512, 256, 0, stream>>>(wl1, 4096 * 1600, acc + 2);
  k_absmax<<<64,  256, 0, stream>>>(wl2, 40960,       acc + 3);

  k_quant_x<<<3136, 256, 0, stream>>>(x, s_in, qx, 802816);
  k_quant_w<<<2, 256, 0, stream>>>(w1, 288, acc + 0, sbuf + 0, qw1);
  k_quant_w2p<<<80, 256, 0, stream>>>(w2, acc + 1, sbuf + 1, qw2p);
  k_quant_w<<<(6553600 + 255) / 256, 256, 0, stream>>>(wl1, 6553600, acc + 2, sbuf + 2, qwl1);
  k_quant_w<<<160, 256, 0, stream>>>(wl2, 40960, acc + 3, sbuf + 3, qwl2);

  k_conv1<<<21632, 256, 0, stream>>>(qx, qw1, sbuf, s_in, s_a1, act1);

  k_conv2_wmma<<<968, 256, 0, stream>>>(act1, qw2p, sbuf, s_in, s_a2, raw3);
  k_pool2<<<6400, 256, 0, stream>>>(raw3, s_in, s_a2, act2);

  k_fc1_wmma<<<512, 256, 0, stream>>>(act2, qwl1, sbuf, s_in, s_a3, act3);

  k_fc2<<<40, 256, 0, stream>>>(act3, qwl2, sbuf, s_a3, (float*)d_out);
}
